// MultiHeadSelfAttention_79731772883647
// MI455X (gfx1250) — matrix operand order repaired and checked by reading, not compiled, not run on hardware
//
#include <hip/hip_runtime.h>

// ---------------------------------------------------------------------------
// MultiHeadSelfAttention with 3D RoPE, flash-attention style, for gfx1250.
// B=2, S=2048, D_MODEL=1024, N_HEADS=16, HEAD_DIM=64.
// All matmuls use v_wmma_f32_16x16x32_f16 (f32 accumulate).
// ---------------------------------------------------------------------------

static constexpr int B_   = 2;
static constexpr int S_   = 2048;
static constexpr int DM_  = 1024;
static constexpr int H_   = 16;
static constexpr int DH_  = 64;
static constexpr int E3_  = 3 * DM_;   // 3072

typedef __attribute__((ext_vector_type(4)))  _Float16 v4h;
typedef __attribute__((ext_vector_type(8)))  _Float16 v8h;
typedef __attribute__((ext_vector_type(16))) _Float16 v16h;
typedef __attribute__((ext_vector_type(8)))  float    v8f;

__device__ __forceinline__ v8f wmma_f16(v16h a, v16h b, v8f c) {
  // D = A(16xK=32, f16) x B(32x16, f16) + C(16x16, f32)
  return __builtin_amdgcn_wmma_f32_16x16x32_f16(false, a, false, b,
                                                (short)0, c, false, false);
}

// A-operand fragment (16x32 f16): per ISA, lane holds row (lane&15), K values
// {g*8 .. g*8+7} and {g*8+16 .. g*8+23} with g = lane>>4 -> two 16B loads.
__device__ __forceinline__ v16h ld_a_frag(const _Float16* row, int off) {
  v8h lo = *(const v8h*)(row + off);
  v8h hi = *(const v8h*)(row + off + 16);
  return __builtin_shufflevector(lo, hi, 0, 1, 2, 3, 4, 5, 6, 7,
                                         8, 9, 10, 11, 12, 13, 14, 15);
}

// ---------------------------------------------------------------------------
// f32 -> f16 cast, 4 elements per thread
// ---------------------------------------------------------------------------
__global__ void cast_f32_to_f16_x4(const float* __restrict__ in,
                                   _Float16* __restrict__ out, int n4) {
  int i = blockIdx.x * blockDim.x + threadIdx.x;
  if (i < n4) {
    float4 v = ((const float4*)in)[i];
    v4h o = {(_Float16)v.x, (_Float16)v.y, (_Float16)v.z, (_Float16)v.w};
    ((v4h*)out)[i] = o;
  }
}

// ---------------------------------------------------------------------------
// C(MxN) = A(MxK) * W(NxK)^T, f16 inputs, f32 accumulate, OutT output.
// 256 threads = 8 waves; wave tile 32(M) x 64(N); block tile 64 x 256.
// Unroll-by-2 software pipeline with two named fragment buffers (no copies).
// Requires M%64==0, N%256==0, K%64==0 (true for all uses here).
// ---------------------------------------------------------------------------
template <typename OutT>
__global__ __launch_bounds__(256) void gemm_f16_nt(
    const _Float16* __restrict__ A, const _Float16* __restrict__ W,
    OutT* __restrict__ C, int M, int N, int K) {
  const int lane = threadIdx.x & 31;
  const int wave = threadIdx.x >> 5;
  const int col  = lane & 15;
  const int grp  = lane >> 4;
  const int m0 = blockIdx.y * 64 + (wave >> 2) * 32;
  const int n0 = blockIdx.x * 256 + (wave & 3) * 64;

  const _Float16* arow0 = A + (long)(m0 + col) * K;
  const _Float16* arow1 = arow0 + (long)16 * K;
  const _Float16* wrow  = W + (long)(n0 + col) * K;

  v8f acc[2][4] = {};

  // buffer set A: k=0
  v16h af0a = ld_a_frag(arow0, grp * 8);
  v16h af1a = ld_a_frag(arow1, grp * 8);
  v16h bfa[4];
#pragma unroll
  for (int t = 0; t < 4; ++t) bfa[t] = ld_a_frag(wrow + (long)t * 16 * K, grp * 8);
  v16h af0b, af1b, bfb[4];

  for (int k = 0; k < K; k += 64) {
    // prefetch k+32 into buffer set B (always in-range: K % 64 == 0)
    {
      const int kn = k + 32;
      af0b = ld_a_frag(arow0 + kn, grp * 8);
      af1b = ld_a_frag(arow1 + kn, grp * 8);
#pragma unroll
      for (int t = 0; t < 4; ++t)
        bfb[t] = ld_a_frag(wrow + (long)t * 16 * K + kn, grp * 8);
    }
#pragma unroll
    for (int t = 0; t < 4; ++t) {
      acc[0][t] = wmma_f16(af0a, bfa[t], acc[0][t]);
      acc[1][t] = wmma_f16(af1a, bfa[t], acc[1][t]);
    }
    if (k + 64 < K) {  // prefetch k+64 into buffer set A (scalar condition)
      const int kn = k + 64;
      af0a = ld_a_frag(arow0 + kn, grp * 8);
      af1a = ld_a_frag(arow1 + kn, grp * 8);
#pragma unroll
      for (int t = 0; t < 4; ++t)
        bfa[t] = ld_a_frag(wrow + (long)t * 16 * K + kn, grp * 8);
    }
#pragma unroll
    for (int t = 0; t < 4; ++t) {
      acc[0][t] = wmma_f16(af0b, bfb[t], acc[0][t]);
      acc[1][t] = wmma_f16(af1b, bfb[t], acc[1][t]);
    }
  }

#pragma unroll
  for (int s = 0; s < 2; ++s) {
#pragma unroll
    for (int t = 0; t < 4; ++t) {
#pragma unroll
      for (int r = 0; r < 8; ++r) {
        C[(long)(m0 + s * 16 + grp * 8 + r) * N + n0 + t * 16 + col] =
            (OutT)acc[s][t][r];
      }
    }
  }
}

// ---------------------------------------------------------------------------
// 3D RoPE + scatter: qkv[B,S,3,H,DH] (f16) -> Q,K in [B,H,S,DH], V transposed
// into [B,H,DH,S] (so the P*V WMMA B-operand is a contiguous load).
// One thread per (b, s, h, pair): 2^21 threads. D_X=20, D_Y=20, D_Z=24.
// ---------------------------------------------------------------------------
__global__ void rope_scatter_kernel(const _Float16* __restrict__ qkv,
                                    const int* __restrict__ pos,
                                    _Float16* __restrict__ Qo,
                                    _Float16* __restrict__ Ko,
                                    _Float16* __restrict__ VTo) {
  int idx = blockIdx.x * blockDim.x + threadIdx.x;
  const int p = idx & 31;
  const int h = (idx >> 5) & 15;
  const int s = (idx >> 9) & 2047;
  const int b = idx >> 20;

  int seg, j, dseg, doff;
  if (p < 10)      { seg = 0; j = p;      dseg = 20; doff = 0;  }
  else if (p < 20) { seg = 1; j = p - 10; dseg = 20; doff = 20; }
  else             { seg = 2; j = p - 20; dseg = 24; doff = 40; }
  const int d0 = doff + 2 * j;

  const float ang = (float)pos[((long)b * S_ + s) * 3 + seg] *
                    __expf(-(2.0f * (float)j / (float)dseg) * 9.210340371976184f);
  const float c = __cosf(ang), sn = __sinf(ang);

  const long rbase = ((long)b * S_ + s) * E3_ + h * DH_ + d0;
  const float qx = (float)qkv[rbase];        const float qy = (float)qkv[rbase + 1];
  const float kx = (float)qkv[rbase + 1024]; const float ky = (float)qkv[rbase + 1025];
  const float vx = (float)qkv[rbase + 2048]; const float vy = (float)qkv[rbase + 2049];

  const long o = ((long)(b * H_ + h) * S_ + s) * DH_ + d0;
  Qo[o]     = (_Float16)(qx * c - qy * sn);
  Qo[o + 1] = (_Float16)(qy * c + qx * sn);
  Ko[o]     = (_Float16)(kx * c - ky * sn);
  Ko[o + 1] = (_Float16)(ky * c + kx * sn);

  const long ov = ((long)(b * H_ + h) * DH_ + d0) * S_ + s;
  VTo[ov]      = (_Float16)vx;
  VTo[ov + S_] = (_Float16)vy;
}

// ---------------------------------------------------------------------------
// Flash attention. Block = 8 waves; wave owns a 16-query tile, iterates key
// chunks of 32. Computes S^T = K_tile x Q^T so the f32 C-layout of the score
// tile IS the A-layout of P for the P*V WMMA (zero-shuffle handoff).
// qbase / ballot mask are readfirstlane'd so all loop/branch conditions are
// SGPR-uniform (scalar branches, EXEC untouched, WMMA-legal everywhere).
// Unroll-by-2 double buffering: no fragment copies between iterations.
// ---------------------------------------------------------------------------
struct Chunk {
  v16h a00, a01, a10, a11;   // K-tile A-fragments (2 key sub-tiles x 2 hd chunks)
  v16h vf0, vf1, vf2, vf3;   // V^T B-fragments (4 hd tiles)
  int  kbits;                // attention-mask bits for the 32 keys (uniform)
};

__device__ __forceinline__ Chunk load_chunk(const _Float16* __restrict__ Kb,
                                            const _Float16* __restrict__ Vb,
                                            const unsigned char* __restrict__ am,
                                            int kbase, int lane, int col, int grp) {
  Chunk c;
  const _Float16* k0p = Kb + (long)(kbase + col) * DH_;
  const _Float16* k1p = k0p + 16 * DH_;
  c.a00 = ld_a_frag(k0p,      grp * 8);
  c.a01 = ld_a_frag(k0p + 32, grp * 8);
  c.a10 = ld_a_frag(k1p,      grp * 8);
  c.a11 = ld_a_frag(k1p + 32, grp * 8);
  const _Float16* vcol = Vb + (long)col * S_ + kbase;
  c.vf0 = ld_a_frag(vcol, grp * 8);
  c.vf1 = ld_a_frag(vcol + 16 * S_, grp * 8);
  c.vf2 = ld_a_frag(vcol + 32 * S_, grp * 8);
  c.vf3 = ld_a_frag(vcol + 48 * S_, grp * 8);
  c.kbits = __builtin_amdgcn_readfirstlane(
      (int)(unsigned)__ballot((int)(am[kbase + lane] != 0)));
  return c;
}

__device__ __forceinline__ void process_chunk(
    const Chunk& c, int kbase, int qbase_u, int qcol, int grp,
    v16h qf0, v16h qf1, v8f (&acc)[4], float& m_run, float& l_run) {
  const float scale = 0.125f;  // HEAD_DIM^-0.5

  // ---- S^T = K_tile x Q^T ----
  v8f st0 = {}, st1 = {};
  st0 = wmma_f16(c.a00, qf0, st0);
  st0 = wmma_f16(c.a01, qf1, st0);
  st1 = wmma_f16(c.a10, qf0, st1);
  st1 = wmma_f16(c.a11, qf1, st1);

  // ---- mask + scale (scalar-branch fast path for fully-valid chunks) ----
  float s0[8], s1[8];
  float cm = -1e30f;
  if (c.kbits == -1 && kbase + 31 <= qbase_u) {  // SGPR-uniform condition
#pragma unroll
    for (int r = 0; r < 8; ++r) {
      s0[r] = st0[r] * scale;
      s1[r] = st1[r] * scale;
      cm = fmaxf(cm, fmaxf(s0[r], s1[r]));
    }
  } else {
#pragma unroll
    for (int r = 0; r < 8; ++r) {
      const int krow = 8 * grp + r;
      const int key0 = kbase + krow;
      const int key1 = key0 + 16;
      const bool ok0 = (key0 <= qcol) && (((unsigned)c.kbits >> krow) & 1u);
      const bool ok1 = (key1 <= qcol) && (((unsigned)c.kbits >> (krow + 16)) & 1u);
      s0[r] = ok0 ? st0[r] * scale : -1e30f;
      s1[r] = ok1 ? st1[r] * scale : -1e30f;
      cm = fmaxf(cm, fmaxf(s0[r], s1[r]));
    }
  }
  cm = fmaxf(cm, __shfl_xor(cm, 16));
  const float m_new = fmaxf(m_run, cm);
  const float alpha = __expf(m_run - m_new);

  // ---- exponentiate -> P fragment (C-layout of S^T == A-layout of P) ----
  v16h pf;
  float rowsum = 0.0f;
#pragma unroll
  for (int r = 0; r < 8; ++r) {
    const float p0 = __expf(s0[r] - m_new);
    const float p1 = __expf(s1[r] - m_new);
    rowsum += p0 + p1;
    pf[r]     = (_Float16)p0;
    pf[r + 8] = (_Float16)p1;
  }
  rowsum += __shfl_xor(rowsum, 16);
  l_run = l_run * alpha + rowsum;
  m_run = m_new;

  // ---- rescale accumulators (alpha for row q lives in lane q-qbase) ----
#pragma unroll
  for (int r = 0; r < 8; ++r) {
    const float ar = __shfl(alpha, 8 * grp + r);
#pragma unroll
    for (int t = 0; t < 4; ++t) acc[t][r] *= ar;
  }

  // ---- P * V over the 32-key chunk ----
  acc[0] = wmma_f16(pf, c.vf0, acc[0]);
  acc[1] = wmma_f16(pf, c.vf1, acc[1]);
  acc[2] = wmma_f16(pf, c.vf2, acc[2]);
  acc[3] = wmma_f16(pf, c.vf3, acc[3]);
}

__global__ __launch_bounds__(256) void flash_attn_kernel(
    const _Float16* __restrict__ Q, const _Float16* __restrict__ Kt,
    const _Float16* __restrict__ VT, const unsigned char* __restrict__ amask,
    _Float16* __restrict__ O) {
  const int lane = threadIdx.x & 31;
  const int wave = threadIdx.x >> 5;
  const int col  = lane & 15;   // query column in score tile / hd column in out
  const int grp  = lane >> 4;   // 0/1
  const int b    = blockIdx.z;
  const int h    = blockIdx.y;
  // wave-uniform by construction; readfirstlane makes the compiler see it
  const int qbase = __builtin_amdgcn_readfirstlane(blockIdx.x * 128 + wave * 16);

  const long bh = (long)(b * H_ + h);
  const _Float16* Qb = Q  + bh * S_ * DH_;
  const _Float16* Kb = Kt + bh * S_ * DH_;
  const _Float16* Vb = VT + bh * DH_ * S_;
  const unsigned char* am = amask + (long)b * S_;

  // Q^T B-operand fragments, head-dim chunks [0,32) and [32,64)
  const _Float16* qrow = Qb + (long)(qbase + col) * DH_;
  const v16h qf0 = ld_a_frag(qrow, grp * 8);
  const v16h qf1 = ld_a_frag(qrow + 32, grp * 8);

  v8f acc[4] = {};
  float m_run = -1e30f, l_run = 0.0f;
  const int qcol = qbase + col;

  const int kend = qbase + 16;  // causal bound (SGPR-uniform)
  Chunk c0 = load_chunk(Kb, Vb, am, 0, lane, col, grp);
  Chunk c1;
  int kbase = 0;
  for (;;) {
    if (kbase + 32 < kend) c1 = load_chunk(Kb, Vb, am, kbase + 32, lane, col, grp);
    process_chunk(c0, kbase, qbase, qcol, grp, qf0, qf1, acc, m_run, l_run);
    kbase += 32;
    if (kbase >= kend) break;

    if (kbase + 32 < kend) c0 = load_chunk(Kb, Vb, am, kbase + 32, lane, col, grp);
    process_chunk(c1, kbase, qbase, qcol, grp, qf0, qf1, acc, m_run, l_run);
    kbase += 32;
    if (kbase >= kend) break;
  }

  // ---- normalize by softmax denominator and store to [B, S, D_MODEL] ----
  float linv[8];
#pragma unroll
  for (int r = 0; r < 8; ++r) linv[r] = 1.0f / __shfl(l_run, 8 * grp + r);

  _Float16* ob = O + (long)b * S_ * DM_ + h * DH_;
#pragma unroll
  for (int r = 0; r < 8; ++r) {
    const long row = qbase + 8 * grp + r;
#pragma unroll
    for (int t = 0; t < 4; ++t) {
      ob[row * DM_ + t * 16 + col] = (_Float16)(acc[t][r] * linv[r]);
    }
  }
}

// ---------------------------------------------------------------------------
// Launcher
// ---------------------------------------------------------------------------
extern "C" void kernel_launch(void* const* d_in, const int* in_sizes, int n_in,
                              void* d_out, int out_size, void* d_ws, size_t ws_size,
                              hipStream_t stream) {
  (void)in_sizes; (void)n_in; (void)out_size; (void)ws_size;
  const float* hs            = (const float*)d_in[0];          // (B,S,D) f32
  const float* wqkv          = (const float*)d_in[1];          // (3D,D) f32
  const float* wout          = (const float*)d_in[2];          // (D,D) f32
  const unsigned char* amask = (const unsigned char*)d_in[3];  // (B,S) bool
  // d_in[4] (causal_mask) computed analytically
  const int* pos             = (const int*)d_in[5];            // (B,S,3) i32
  float* out = (float*)d_out;

  char* ws = (char*)d_ws;
  const size_t MB = 1u << 20;
  _Float16* h16    = (_Float16*)(ws + 0);        //  8 MB hidden f16
  _Float16* wq16   = (_Float16*)(ws + 8 * MB);   //  6 MB w_qkv f16
  _Float16* wo16   = (_Float16*)(ws + 14 * MB);  //  2 MB w_out f16
  _Float16* qkv16  = (_Float16*)(ws + 16 * MB);  // 24 MB qkv f16
  _Float16* q16    = (_Float16*)(ws + 40 * MB);  //  8 MB (B,H,S,DH)
  _Float16* k16    = (_Float16*)(ws + 48 * MB);  //  8 MB (B,H,S,DH)
  _Float16* vt16   = (_Float16*)(ws + 56 * MB);  //  8 MB (B,H,DH,S)
  _Float16* attn16 = h16;                        //  reuse: hidden dead by then

  const int NHS = B_ * S_ * DM_;       // 4194304
  const int NWQ = E3_ * DM_;           // 3145728
  const int NWO = DM_ * DM_;           // 1048576
  cast_f32_to_f16_x4<<<NHS / 4 / 256, 256, 0, stream>>>(hs, h16, NHS / 4);
  cast_f32_to_f16_x4<<<NWQ / 4 / 256, 256, 0, stream>>>(wqkv, wq16, NWQ / 4);
  cast_f32_to_f16_x4<<<NWO / 4 / 256, 256, 0, stream>>>(wout, wo16, NWO / 4);

  // QKV projection: (4096 x 1024) x (3072 x 1024)^T -> 4096 x 3072 (f16)
  gemm_f16_nt<_Float16><<<dim3(E3_ / 256, (B_ * S_) / 64), 256, 0, stream>>>(
      h16, wq16, qkv16, B_ * S_, E3_, DM_);

  // RoPE + scatter into per-head layouts
  rope_scatter_kernel<<<(B_ * S_ * H_ * 32) / 256, 256, 0, stream>>>(
      qkv16, pos, q16, k16, vt16);

  // Flash attention: grid (S/128 q-tiles, heads, batch)
  flash_attn_kernel<<<dim3(S_ / 128, H_, B_), 256, 0, stream>>>(
      q16, k16, vt16, amask, attn16);

  // Output projection: (4096 x 1024) x (1024 x 1024)^T -> f32 d_out
  gemm_f16_nt<float><<<dim3(DM_ / 256, (B_ * S_) / 64), 256, 0, stream>>>(
      attn16, wo16, out, B_ * S_, DM_, DM_);
}
